// GraphBased_4054449128250
// MI455X (gfx1250) — matrix operand order repaired and checked by reading, not compiled, not run on hardware
//
#include <hip/hip_runtime.h>
#include <hip/hip_bf16.h>
#include <cstdint>

typedef _Float16 half_t;
typedef __attribute__((ext_vector_type(16))) _Float16 v16h;
typedef __attribute__((ext_vector_type(8)))  _Float16 h8;
typedef __attribute__((ext_vector_type(8)))  float    v8f;

#define NI   4096     // nodes
#define LF   500      // feature dim
#define LP   512      // padded feature dim (mult of 32)
#define NC   5        // classes / coarse nodes

// ---------------------------------------------------------------------------
// Templated WMMA GEMM:  C[M,N] = X0 @ Y0^T (+ X1 @ Y1^T)  (+bias, activation)
// X,Y row-major f16, K multiple of 32 (pre-padded with zeros).
// Block: 256 threads = 8 waves (2 M-waves x 4 N-waves); block tile 128x128,
// wave tile 64x32 (4x2 accumulators of v8f), K step 32 via LDS.
// MODE 0: store f32.  MODE 1: +bias, ReLU.  MODE 2: dual K pass, +bias, LeakyReLU(0.01)
// ---------------------------------------------------------------------------
template<int MODE>
__global__ __launch_bounds__(256) void wmma_gemm(
    const half_t* __restrict__ X0, int ldx0,
    const half_t* __restrict__ Y0, int ldy0, int K0,
    const half_t* __restrict__ X1, int ldx1,
    const half_t* __restrict__ Y1, int ldy1, int K1,
    int Mreal, int Nreal,
    const float* __restrict__ bias,
    float* __restrict__ C, int ldc)
{
  constexpr int LDT = 40;                       // padded LDS row (halves)
  __shared__ half_t As[128 * LDT];
  __shared__ half_t Bs[128 * LDT];

  const int tid  = threadIdx.x;
  const int lane = tid & 31;
  const int wave = tid >> 5;
  const int mw   = wave & 1;                    // 0..1
  const int nw   = wave >> 1;                   // 0..3
  const int bm   = blockIdx.x * 128;
  const int bn   = blockIdx.y * 128;

  v8f acc[4][2];
  #pragma unroll
  for (int mi = 0; mi < 4; ++mi)
    #pragma unroll
    for (int ni = 0; ni < 2; ++ni)
      #pragma unroll
      for (int q = 0; q < 8; ++q) acc[mi][ni][q] = 0.0f;

  // per-lane fragment constants (ISA 7.12.2 layouts, wave32)
  const int lm = lane & 15;
  const int ak = (lane < 16) ? 0 : 8;           // A fragment K base
  const int bk = (lane < 16) ? 0 : 8;           // B fragment K base
  const int lr = tid >> 1;                      // LDS fill: row 0..127
  const int lc = (tid & 1) << 4;                // LDS fill: half-col 0 or 16

  const int npasses = (MODE == 2) ? 2 : 1;
  for (int pass = 0; pass < npasses; ++pass) {
    const half_t* X  = pass ? X1 : X0;
    const half_t* Y  = pass ? Y1 : Y0;
    const int     ldx = pass ? ldx1 : ldx0;
    const int     ldy = pass ? ldy1 : ldy0;
    const int     K   = pass ? K1  : K0;

    for (int k0 = 0; k0 < K; k0 += 32) {
      __syncthreads();
      {   // stage A tile 128x32
        h8 v0, v1;
        int gr = bm + lr;
        if (gr < Mreal) {
          const h8* s = (const h8*)(X + (size_t)gr * ldx + k0 + lc);
          v0 = s[0]; v1 = s[1];
        } else {
          #pragma unroll
          for (int q = 0; q < 8; ++q) { v0[q] = (half_t)0; v1[q] = (half_t)0; }
        }
        h8* d = (h8*)(As + lr * LDT + lc);
        d[0] = v0; d[1] = v1;
      }
      {   // stage B tile (rows = N index, cols = K) 128x32
        h8 v0, v1;
        int gr = bn + lr;
        if (gr < Nreal) {
          const h8* s = (const h8*)(Y + (size_t)gr * ldy + k0 + lc);
          v0 = s[0]; v1 = s[1];
        } else {
          #pragma unroll
          for (int q = 0; q < 8; ++q) { v0[q] = (half_t)0; v1[q] = (half_t)0; }
        }
        h8* d = (h8*)(Bs + lr * LDT + lc);
        d[0] = v0; d[1] = v1;
      }
      __syncthreads();

      #pragma unroll
      for (int mi = 0; mi < 4; ++mi) {
        v16h a;
        const half_t* ap = As + (mw * 64 + mi * 16 + lm) * LDT + ak;
        ((h8*)&a)[0] = *(const h8*)(ap);        // K = ak + 0..7
        ((h8*)&a)[1] = *(const h8*)(ap + 16);   // K = ak + 16..23
        #pragma unroll
        for (int ni = 0; ni < 2; ++ni) {
          v16h b;
          const half_t* bp = Bs + (nw * 32 + ni * 16 + lm) * LDT + bk;
          ((h8*)&b)[0] = *(const h8*)(bp);      // K = bk + 0..7
          ((h8*)&b)[1] = *(const h8*)(bp + 16); // K = bk + 16..23
          acc[mi][ni] = __builtin_amdgcn_wmma_f32_16x16x32_f16(
              false, a, false, b, (short)0, acc[mi][ni], false, false);
        }
      }
    }
  }

  const int rb = (lane < 16) ? 0 : 8;
  #pragma unroll
  for (int mi = 0; mi < 4; ++mi)
    #pragma unroll
    for (int ni = 0; ni < 2; ++ni)
      #pragma unroll
      for (int q = 0; q < 8; ++q) {
        int row = bm + mw * 64 + mi * 16 + rb + q;
        int col = bn + nw * 32 + ni * 16 + lm;
        if (row < Mreal && col < Nreal) {
          float v = acc[mi][ni][q];
          if (MODE == 1) { v += bias[col]; v = fmaxf(v, 0.0f); }
          if (MODE == 2) { v += bias[col]; v = (v > 0.0f) ? v : 0.01f * v; }
          C[(size_t)row * ldc + col] = v;
        }
      }
}

// ---------------------------------------------------------------------------
// conv1 5x5 valid + ReLU + maxpool2  : [N,1,28,28] -> [N,20,12,12]
// ---------------------------------------------------------------------------
__global__ __launch_bounds__(256) void conv1_kernel(
    const float* __restrict__ x, const float* __restrict__ w,
    const float* __restrict__ b, float* __restrict__ out)
{
  int idx = blockIdx.x * 256 + threadIdx.x;
  if (idx >= NI * 20 * 144) return;
  int px = idx % 12, py = (idx / 12) % 12, c = (idx / 144) % 20, n = idx / 2880;
  const float* xi = x + (size_t)n * 784;
  const float* wc = w + c * 25;
  float bias = b[c], mx = -INFINITY;
  #pragma unroll
  for (int dy = 0; dy < 2; ++dy)
    #pragma unroll
    for (int dx = 0; dx < 2; ++dx) {
      int oy = 2 * py + dy, ox = 2 * px + dx;
      float s = bias;
      #pragma unroll
      for (int ky = 0; ky < 5; ++ky)
        #pragma unroll
        for (int kx = 0; kx < 5; ++kx)
          s += xi[(oy + ky) * 28 + ox + kx] * wc[ky * 5 + kx];
      mx = fmaxf(mx, fmaxf(s, 0.0f));
    }
  out[idx] = mx;
}

// ---------------------------------------------------------------------------
// conv2 5x5 valid + ReLU + maxpool2 : [N,20,12,12] -> hflat f16 [N,800]
// ---------------------------------------------------------------------------
__global__ __launch_bounds__(256) void conv2_kernel(
    const float* __restrict__ c1, const float* __restrict__ w,
    const float* __restrict__ b, half_t* __restrict__ hflat)
{
  __shared__ float tile[20 * 144];
  int n = blockIdx.x;
  const float* src = c1 + (size_t)n * 2880;
  for (int i = threadIdx.x; i < 2880; i += 256) tile[i] = src[i];
  __syncthreads();
  for (int o = threadIdx.x; o < 800; o += 256) {
    int c = o / 16, py = (o % 16) / 4, px = o % 4;
    const float* wc = w + c * 500;
    float mx = -INFINITY;
    for (int dy = 0; dy < 2; ++dy)
      for (int dx = 0; dx < 2; ++dx) {
        int oy = 2 * py + dy, ox = 2 * px + dx;
        float s = b[c];
        for (int ci = 0; ci < 20; ++ci) {
          const float* tp = &tile[ci * 144 + oy * 12 + ox];
          const float* wp = &wc[ci * 25];
          #pragma unroll
          for (int ky = 0; ky < 5; ++ky)
            #pragma unroll
            for (int kx = 0; kx < 5; ++kx)
              s += tp[ky * 12 + kx] * wp[ky * 5 + kx];
        }
        mx = fmaxf(mx, fmaxf(s, 0.0f));
      }
    hflat[(size_t)n * 800 + o] = (half_t)mx;
  }
}

// f32 -> f16 with K padding (pad cols written as 0)
__global__ __launch_bounds__(256) void cvt_pad_f16(
    const float* __restrict__ src, half_t* __restrict__ dst,
    int rows, int cols, int dcols)
{
  size_t idx = (size_t)blockIdx.x * 256 + threadIdx.x;
  if (idx >= (size_t)rows * dcols) return;
  int r = (int)(idx / dcols), c = (int)(idx % dcols);
  dst[idx] = (c < cols) ? (half_t)src[(size_t)r * cols + c] : (half_t)0;
}

// H [n,L] f32 -> H^T f16 [Lp, n] (rows >= L zero)
__global__ __launch_bounds__(256) void cvt_T_f16(
    const float* __restrict__ H, half_t* __restrict__ HT, int n, int L, int Lp)
{
  size_t idx = (size_t)blockIdx.x * 256 + threadIdx.x;
  if (idx >= (size_t)Lp * n) return;
  int c = (int)(idx / n), j = (int)(idx % n);
  HT[idx] = (c < L) ? (half_t)H[(size_t)j * L + c] : (half_t)0;
}

__global__ void diag_kernel(const float* __restrict__ G, float* __restrict__ sq, int n) {
  int i = blockIdx.x * 256 + threadIdx.x;
  if (i < n) sq[i] = G[(size_t)i * n + i];
}

__global__ void zero_ints(int* p, int n) {
  int i = blockIdx.x * 256 + threadIdx.x;
  if (i < n) p[i] = 0;
}

// histogram of pairwise distances into 32 threshold bins (thr_k = 2.5 + 0.5k)
__global__ __launch_bounds__(256) void hist_kernel(
    const float* __restrict__ G, const float* __restrict__ sq, int n, int* __restrict__ hist)
{
  __shared__ int h[32];
  if (threadIdx.x < 32) h[threadIdx.x] = 0;
  __syncthreads();
  size_t total = (size_t)n * n;
  for (size_t idx = (size_t)blockIdx.x * 256 + threadIdx.x; idx < total;
       idx += (size_t)gridDim.x * 256) {
    int i = (int)(idx / n), j = (int)(idx % n);
    if (i == j) continue;
    float d2 = fmaxf(sq[i] + sq[j] - 2.0f * G[idx], 0.0f);
    float d  = sqrtf(d2);
    int bin  = (d < 2.5f) ? 0 : ((int)floorf((d - 2.5f) * 2.0f) + 1);
    if (bin < 32) atomicAdd(&h[bin], 1);
  }
  __syncthreads();
  if (threadIdx.x < 32) atomicAdd(&hist[threadIdx.x], h[threadIdx.x]);
}

__global__ void select_thr(const int* __restrict__ hist, float* __restrict__ thr2, int n) {
  float minE = 0.1f * (float)n;
  float cum = 0.0f, thr = 2.5f + 0.5f * 31.0f;
  for (int k = 0; k < 32; ++k) {
    cum += (float)hist[k];
    if (cum >= minE) { thr = 2.5f + 0.5f * (float)k; break; }
  }
  *thr2 = thr * thr;
}

// build adjacency f16 + row degrees
__global__ __launch_bounds__(256) void build_A(
    const float* __restrict__ G, const float* __restrict__ sq,
    const float* __restrict__ thr2p, half_t* __restrict__ A,
    float* __restrict__ deg, int n)
{
  __shared__ int red[256];
  int i = blockIdx.x, t = threadIdx.x;
  float t2 = *thr2p, si = sq[i];
  int cnt = 0;
  for (int j = t; j < n; j += 256) {
    float d2 = fmaxf(si + sq[j] - 2.0f * G[(size_t)i * n + j], 0.0f);
    int e = (i != j) && (d2 < t2);
    A[(size_t)i * n + j] = e ? (half_t)1 : (half_t)0;
    cnt += e;
  }
  red[t] = cnt; __syncthreads();
  for (int off = 128; off; off >>= 1) { if (t < off) red[t] += red[t + off]; __syncthreads(); }
  if (t == 0) deg[i] = (float)red[0];
}

// neigh (f16, padded) + ClusterGCN agg (f32) from AH
__global__ __launch_bounds__(256) void neigh_kernel(
    const float* __restrict__ AH, const float* __restrict__ H,
    const float* __restrict__ deg, half_t* __restrict__ neighh, float* __restrict__ agg)
{
  size_t idx = (size_t)blockIdx.x * 256 + threadIdx.x;
  if (idx >= (size_t)NI * LP) return;
  int i = (int)(idx / LP), c = (int)(idx % LP);
  if (c < LF) {
    float ah = AH[(size_t)i * LF + c], d = deg[i];
    neighh[idx] = (half_t)(ah / fmaxf(d, 1.0f));
    agg[(size_t)i * LF + c] = (ah + H[(size_t)i * LF + c]) / (d + 1.0f);
  } else {
    neighh[idx] = (half_t)0;
  }
}

// column batch-norm stats (biased var) over `rows` rows
__global__ __launch_bounds__(256) void bn_stats(
    const float* __restrict__ Z, int rows, int cols,
    float* __restrict__ mean, float* __restrict__ inv)
{
  __shared__ float s1[256], s2[256];
  int c = blockIdx.x, t = threadIdx.x;
  float a = 0.0f, b = 0.0f;
  for (int r = t; r < rows; r += 256) {
    float v = Z[(size_t)r * cols + c]; a += v; b += v * v;
  }
  s1[t] = a; s2[t] = b; __syncthreads();
  for (int off = 128; off; off >>= 1) {
    if (t < off) { s1[t] += s1[t + off]; s2[t] += s2[t + off]; }
    __syncthreads();
  }
  if (t == 0) {
    float m = s1[0] / rows, v = s2[0] / rows - m * m;
    mean[c] = m; inv[c] = rsqrtf(v + 1e-5f);
  }
}

__global__ __launch_bounds__(256) void bn_apply(
    float* __restrict__ Z, const float* __restrict__ mean, const float* __restrict__ inv,
    const float* __restrict__ g, const float* __restrict__ b)
{
  size_t idx = (size_t)blockIdx.x * 256 + threadIdx.x;
  if (idx >= (size_t)NI * LF) return;
  int c = (int)(idx % LF);
  Z[idx] = (Z[idx] - mean[c]) * inv[c] * g[c] + b[c];
}

// S = softmax(agg @ wout^T + bout + H @ wroot^T)  per node
__global__ __launch_bounds__(256) void s_kernel(
    const float* __restrict__ agg, const float* __restrict__ H,
    const float* __restrict__ wout, const float* __restrict__ bout,
    const float* __restrict__ wroot, float* __restrict__ S)
{
  __shared__ float red[256];
  __shared__ float sl[NC];
  int i = blockIdx.x, t = threadIdx.x;
  float p[NC] = {0, 0, 0, 0, 0};
  for (int f = t; f < LF; f += 256) {
    float av = agg[(size_t)i * LF + f], hv = H[(size_t)i * LF + f];
    #pragma unroll
    for (int c = 0; c < NC; ++c) p[c] += av * wout[c * LF + f] + hv * wroot[c * LF + f];
  }
  for (int c = 0; c < NC; ++c) {
    red[t] = p[c]; __syncthreads();
    for (int off = 128; off; off >>= 1) { if (t < off) red[t] += red[t + off]; __syncthreads(); }
    if (t == 0) sl[c] = red[0] + bout[c];
    __syncthreads();
  }
  if (t == 0) {
    float mx = -INFINITY;
    for (int c = 0; c < NC; ++c) mx = fmaxf(mx, sl[c]);
    float e[NC], se = 0.0f;
    for (int c = 0; c < NC; ++c) { e[c] = expf(sl[c] - mx); se += e[c]; }
    for (int c = 0; c < NC; ++c) S[(size_t)i * NC + c] = e[c] / se;
  }
}

// AS = A @ S  (per-row reduction)
__global__ __launch_bounds__(256) void as_kernel(
    const half_t* __restrict__ A, const float* __restrict__ S, float* __restrict__ AS, int n)
{
  __shared__ float red[256];
  int i = blockIdx.x, t = threadIdx.x;
  float p[NC] = {0, 0, 0, 0, 0};
  const half_t* Ar = A + (size_t)i * n;
  for (int j = t; j < n; j += 256) {
    float a = (float)Ar[j];
    #pragma unroll
    for (int c = 0; c < NC; ++c) p[c] += a * S[(size_t)j * NC + c];
  }
  for (int c = 0; c < NC; ++c) {
    red[t] = p[c]; __syncthreads();
    for (int off = 128; off; off >>= 1) { if (t < off) red[t] += red[t + off]; __syncthreads(); }
    if (t == 0) AS[(size_t)i * NC + c] = red[0];
    __syncthreads();
  }
}

// coar[c][f] = sum_i S[i][c] * Z[i][f]
__global__ __launch_bounds__(256) void coar_kernel(
    const float* __restrict__ S, const float* __restrict__ Z, float* __restrict__ coar, int n)
{
  __shared__ float red[256];
  int f = blockIdx.x, t = threadIdx.x;
  float p[NC] = {0, 0, 0, 0, 0};
  for (int i = t; i < n; i += 256) {
    float z = Z[(size_t)i * LF + f];
    #pragma unroll
    for (int c = 0; c < NC; ++c) p[c] += S[(size_t)i * NC + c] * z;
  }
  for (int c = 0; c < NC; ++c) {
    red[t] = p[c]; __syncthreads();
    for (int off = 128; off; off >>= 1) { if (t < off) red[t] += red[t + off]; __syncthreads(); }
    if (t == 0) coar[c * LF + f] = red[0];
    __syncthreads();
  }
}

// everything on the 5-node coarse graph, single block
__global__ __launch_bounds__(512) void tail_kernel(
    const float* __restrict__ S, const float* __restrict__ AS, const float* __restrict__ coar,
    const float* __restrict__ wl, const float* __restrict__ bl, const float* __restrict__ wr,
    const float* __restrict__ bng, const float* __restrict__ bnb,
    const float* __restrict__ l1w, const float* __restrict__ l1b,
    const float* __restrict__ l2w, const float* __restrict__ l2b,
    float* __restrict__ out, int n)
{
  __shared__ float red[512];
  __shared__ float Ac[NC * NC], Mc[NC * NC], degc[NC], logits[NC];
  __shared__ float nb[NC * LF];        // neighc, later pooled
  __shared__ float eb[NC * LF];        // sage2 / embd
  __shared__ float h1[NC * 250];
  int t = threadIdx.x;

  // Ac = S^T @ AS
  for (int p = 0; p < NC * NC; ++p) {
    int c = p / NC, cc = p % NC;
    float s = 0.0f;
    for (int i = t; i < n; i += 512) s += S[(size_t)i * NC + c] * AS[(size_t)i * NC + cc];
    red[t] = s; __syncthreads();
    for (int off = 256; off; off >>= 1) { if (t < off) red[t] += red[t + off]; __syncthreads(); }
    if (t == 0) Ac[p] = red[0];
    __syncthreads();
  }
  if (t == 0) {
    float m = 0.0f;
    for (int p = 0; p < NC * NC; ++p) m += Ac[p];
    m /= (float)(NC * NC);
    for (int c = 0; c < NC; ++c) {
      float d = 0.0f;
      for (int cc = 0; cc < NC; ++cc) {
        float v = (Ac[c * NC + cc] >= m && Ac[c * NC + cc] > 0.0f) ? 1.0f : 0.0f;
        Mc[c * NC + cc] = v; d += v;
      }
      degc[c] = fmaxf(d, 1.0f);
    }
  }
  __syncthreads();
  // neighc = (Mc @ coar) / degc
  for (int o = t; o < NC * LF; o += 512) {
    int c = o / LF, f = o % LF;
    float s = 0.0f;
    for (int cc = 0; cc < NC; ++cc) s += Mc[c * NC + cc] * coar[cc * LF + f];
    nb[o] = s / degc[c];
  }
  __syncthreads();
  // sage2 = neighc @ wl^T + coar @ wr^T + bl ; LeakyReLU
  for (int o = t; o < NC * LF; o += 512) {
    int c = o / LF, f = o % LF;
    float s = bl[f];
    for (int k = 0; k < LF; ++k)
      s += nb[c * LF + k] * wl[f * LF + k] + coar[c * LF + k] * wr[f * LF + k];
    eb[o] = (s > 0.0f) ? s : 0.01f * s;
  }
  __syncthreads();
  // BN over the 5 rows
  for (int f = t; f < LF; f += 512) {
    float m = 0.0f;
    for (int c = 0; c < NC; ++c) m += eb[c * LF + f];
    m *= 0.2f;
    float v = 0.0f;
    for (int c = 0; c < NC; ++c) { float d = eb[c * LF + f] - m; v += d * d; }
    v *= 0.2f;
    float inv = rsqrtf(v + 1e-5f);
    for (int c = 0; c < NC; ++c)
      eb[c * LF + f] = (eb[c * LF + f] - m) * inv * bng[f] + bnb[f];
  }
  __syncthreads();
  // max pool over neighbors incl. self  -> nb
  for (int o = t; o < NC * LF; o += 512) {
    int c = o / LF, f = o % LF;
    float mx = -INFINITY;
    for (int cc = 0; cc < NC; ++cc)
      if (Mc[c * NC + cc] > 0.0f || cc == c) mx = fmaxf(mx, eb[cc * LF + f]);
    nb[o] = mx;
  }
  __syncthreads();
  // h1 = leaky(pooled @ l1w^T + l1b)
  for (int o = t; o < NC * 250; o += 512) {
    int c = o / 250, f = o % 250;
    float s = l1b[f];
    for (int k = 0; k < LF; ++k) s += nb[c * LF + k] * l1w[f * LF + k];
    h1[o] = (s > 0.0f) ? s : 0.01f * s;
  }
  __syncthreads();
  if (t < NC) {
    float s = l2b[0];
    for (int k = 0; k < 250; ++k) s += h1[t * 250 + k] * l2w[k];
    logits[t] = (s > 0.0f) ? s : 0.01f * s;
  }
  __syncthreads();
  if (t == 0) {
    float mx = -INFINITY;
    for (int c = 0; c < NC; ++c) mx = fmaxf(mx, logits[c]);
    float se = 0.0f, me = 0.0f;
    for (int c = 0; c < NC; ++c) { float e = expf(logits[c] - mx); se += e; me = fmaxf(me, e); }
    float p = me / se;
    out[0] = p;
    out[1] = (p >= 0.5f) ? 1.0f : 0.0f;
  }
}

// ---------------------------------------------------------------------------
extern "C" void kernel_launch(void* const* d_in, const int* in_sizes, int n_in,
                              void* d_out, int out_size, void* d_ws, size_t ws_size,
                              hipStream_t stream) {
  (void)in_sizes; (void)n_in; (void)out_size; (void)ws_size;
  const float* x        = (const float*)d_in[0];
  const float* c1w      = (const float*)d_in[1];
  const float* c1b      = (const float*)d_in[2];
  const float* c2w      = (const float*)d_in[3];
  const float* c2b      = (const float*)d_in[4];
  const float* fcw      = (const float*)d_in[5];
  const float* fcb      = (const float*)d_in[6];
  const float* sage_wl  = (const float*)d_in[7];
  const float* sage_bl  = (const float*)d_in[8];
  const float* sage_wr  = (const float*)d_in[9];
  const float* bn_g     = (const float*)d_in[10];
  const float* bn_b     = (const float*)d_in[11];
  const float* cg_wout  = (const float*)d_in[12];
  const float* cg_bout  = (const float*)d_in[13];
  const float* cg_wroot = (const float*)d_in[14];
  const float* l1w      = (const float*)d_in[15];
  const float* l1b      = (const float*)d_in[16];
  const float* l2w      = (const float*)d_in[17];
  const float* l2b      = (const float*)d_in[18];
  float* out = (float*)d_out;

  uint8_t* ws = (uint8_t*)d_ws;
  size_t off = 0;
  auto take = [&](size_t bytes) -> void* {
    void* p = ws + off;
    off += (bytes + 255) & ~(size_t)255;
    return p;
  };
  float*  G      = (float*) take((size_t)NI * NI * 4);   // also conv1 output (47MB < 67MB)
  half_t* A      = (half_t*)take((size_t)NI * NI * 2);
  half_t* Hh     = (half_t*)take((size_t)NI * LP * 2);
  half_t* HhT    = (half_t*)take((size_t)LP * NI * 2);
  half_t* hflat  = (half_t*)take((size_t)NI * 800 * 2);
  half_t* wlh    = (half_t*)take((size_t)LP * LP * 2);
  half_t* wrh    = (half_t*)take((size_t)LP * LP * 2);
  half_t* fwh    = (half_t*)take((size_t)LP * 800 * 2);
  float*  H      = (float*) take((size_t)NI * LF * 4);
  float*  AH     = (float*) take((size_t)NI * LF * 4);
  half_t* neighh = (half_t*)take((size_t)NI * LP * 2);
  float*  agg    = (float*) take((size_t)NI * LF * 4);
  float*  Z      = (float*) take((size_t)NI * LF * 4);
  float*  S      = (float*) take((size_t)NI * NC * 4);
  float*  AS     = (float*) take((size_t)NI * NC * 4);
  float*  deg    = (float*) take((size_t)NI * 4);
  float*  sq     = (float*) take((size_t)NI * 4);
  int*    hist   = (int*)   take(32 * 4);
  float*  thr2   = (float*) take(256);
  float*  cmean  = (float*) take(LP * 4);
  float*  cinv   = (float*) take(LP * 4);
  float*  coar   = (float*) take((size_t)NC * LF * 4);
  float*  c1out  = G;   // alias: dead before G is written

  // 1) CNN feature extractor
  conv1_kernel<<<(NI * 20 * 144 + 255) / 256, 256, 0, stream>>>(x, c1w, c1b, c1out);
  conv2_kernel<<<NI, 256, 0, stream>>>(c1out, c2w, c2b, hflat);
  cvt_pad_f16<<<(500 * 800 + 255) / 256, 256, 0, stream>>>(fcw, fwh, 500, 800, 800);
  wmma_gemm<1><<<dim3(NI / 128, (LF + 127) / 128), 256, 0, stream>>>(
      hflat, 800, fwh, 800, 800,
      (const half_t*)nullptr, 0, (const half_t*)nullptr, 0, 0,
      NI, LF, fcb, H, LF);

  // 2) f16 copies of H
  cvt_pad_f16<<<((size_t)NI * LP + 255) / 256, 256, 0, stream>>>(H, Hh, NI, LF, LP);
  cvt_T_f16<<<((size_t)LP * NI + 255) / 256, 256, 0, stream>>>(H, HhT, NI, LF, LP);

  // 3) Gram matrix G = Hh @ Hh^T  (overwrites conv1 scratch)
  wmma_gemm<0><<<dim3(NI / 128, NI / 128), 256, 0, stream>>>(
      Hh, LP, Hh, LP, LP,
      (const half_t*)nullptr, 0, (const half_t*)nullptr, 0, 0,
      NI, NI, (const float*)nullptr, G, NI);

  // 4) radius-graph threshold + adjacency
  diag_kernel<<<(NI + 255) / 256, 256, 0, stream>>>(G, sq, NI);
  zero_ints<<<1, 256, 0, stream>>>(hist, 32);
  hist_kernel<<<4096, 256, 0, stream>>>(G, sq, NI, hist);
  select_thr<<<1, 1, 0, stream>>>(hist, thr2, NI);
  build_A<<<NI, 256, 0, stream>>>(G, sq, thr2, A, deg, NI);

  // 5) AH = A @ H   (Y = H^T so rows are contiguous)
  wmma_gemm<0><<<dim3(NI / 128, (LF + 127) / 128), 256, 0, stream>>>(
      A, NI, HhT, NI, NI,
      (const half_t*)nullptr, 0, (const half_t*)nullptr, 0, 0,
      NI, LF, (const float*)nullptr, AH, LF);
  neigh_kernel<<<((size_t)NI * LP + 255) / 256, 256, 0, stream>>>(AH, H, deg, neighh, agg);

  // 6) SAGE: Z = leaky(neigh @ wl^T + H @ wr^T + bl), then BN
  cvt_pad_f16<<<(500 * LP + 255) / 256, 256, 0, stream>>>(sage_wl, wlh, LF, LF, LP);
  cvt_pad_f16<<<(500 * LP + 255) / 256, 256, 0, stream>>>(sage_wr, wrh, LF, LF, LP);
  wmma_gemm<2><<<dim3(NI / 128, (LF + 127) / 128), 256, 0, stream>>>(
      neighh, LP, wlh, LP, LP,
      Hh, LP, wrh, LP, LP,
      NI, LF, sage_bl, Z, LF);
  bn_stats<<<LF, 256, 0, stream>>>(Z, NI, LF, cmean, cinv);
  bn_apply<<<((size_t)NI * LF + 255) / 256, 256, 0, stream>>>(Z, cmean, cinv, bn_g, bn_b);

  // 7) cluster assignment, coarsening inputs
  s_kernel<<<NI, 256, 0, stream>>>(agg, H, cg_wout, cg_bout, cg_wroot, S);
  as_kernel<<<NI, 256, 0, stream>>>(A, S, AS, NI);
  coar_kernel<<<LF, 256, 0, stream>>>(S, Z, coar, NI);

  // 8) coarse graph + head
  tail_kernel<<<1, 512, 0, stream>>>(S, AS, coar, sage_wl, sage_bl, sage_wr,
                                     bn_g, bn_b, l1w, l1b, l2w, l2b, out, NI);
}
